// Powen3Retention_44117904064896
// MI455X (gfx1250) — compile-verified, matrix operand order checked by reading, not run on hardware
//
#include <hip/hip_runtime.h>

// ---------------- problem constants ----------------
#define TSEQ   3072
#define HIDDIM 2048
#define NH     16
#define NKV    8
#define HD     128
#define QKV_N  ((NH + 2 * NKV) * HD)   // 4096
#define RMS_EPS  1e-6f
#define NORM_EPS 1e-6f
#define SCALE_Q  0.08838834764831845f  // HD^-0.5
#define LN_THETA_OVER_HALF 0.14391156831212787f // ln(10000)/64

// ---------------- CDNA5 WMMA types ----------------
typedef __attribute__((ext_vector_type(8)))  __bf16 bhalf8;
typedef __attribute__((ext_vector_type(16))) __bf16 bhalf16;
typedef __attribute__((ext_vector_type(8)))  float  float8;

__device__ __forceinline__ float8 f8zero() {
    float8 z = {0.f, 0.f, 0.f, 0.f, 0.f, 0.f, 0.f, 0.f};
    return z;
}

// CDNA5 16-bit A/B fragment: lane l (row/col = l&15) holds K = 8*(l>>4)+{0..7}
// in elements 0..7 and K = 16+8*(l>>4)+{0..7} in elements 8..15.
// -> two contiguous 16-byte loads at p and p+16 elements.
__device__ __forceinline__ bhalf16 ldfrag(const __bf16* p) {
    bhalf8 lo = *(const bhalf8*)p;
    bhalf8 hi = *(const bhalf8*)(p + 16);
    return __builtin_shufflevector(lo, hi, 0, 1, 2, 3, 4, 5, 6, 7,
                                          8, 9, 10, 11, 12, 13, 14, 15);
}

__device__ __forceinline__ float8 wmma_bf16(bhalf16 a, bhalf16 b, float8 c) {
    return __builtin_amdgcn_wmma_f32_16x16x32_bf16(
        /*neg_a=*/false, a, /*neg_b=*/false, b,
        /*c_mod=*/(short)0, c, /*reuse_a=*/false, /*reuse_b=*/false);
}

// ---------------- fp32 -> bf16 pack ----------------
__global__ void cvt_f32_bf16(const float* __restrict__ in,
                             __bf16* __restrict__ out, int n) {
    int i = blockIdx.x * 256 + threadIdx.x;
    if (i < n) out[i] = (__bf16)in[i];
}

// ---------------- generic C = A * B^T GEMM (bf16 in, fp32 out) ----------------
// A: M x K row-major, B: N x K row-major, C: M x N row-major. K % 64 == 0.
// Block = 256 threads = 8 waves arranged 4(row) x 2(col); block tile 128x128.
// Wave tile 32x64: 2 A-frags x 4 B-frags -> 8 WMMAs per 6 fragment loads.
// K loop manually unrolled x2 with ping-pong register sets: no rotation moves,
// loads for the next chunk always in flight behind the current WMMA batch.
__global__ void __launch_bounds__(256)
gemm_bf16_nt(const __bf16* __restrict__ A, const __bf16* __restrict__ B,
             float* __restrict__ C, int M, int N, int K) {
    (void)M;
    const int lane = threadIdx.x & 31;
    const int warp = threadIdx.x >> 5;
    const int lo = lane & 15, hi = lane >> 4;
    const int row0 = blockIdx.y * 128 + (warp & 3) * 32;
    const int col0 = blockIdx.x * 128 + (warp >> 2) * 64;

    float8 acc[2][4];
#pragma unroll
    for (int f = 0; f < 2; ++f)
#pragma unroll
        for (int c = 0; c < 4; ++c) acc[f][c] = f8zero();

    const __bf16* a0 = A + (size_t)(row0 + lo) * K + 8 * hi;
    const __bf16* a1 = a0 + (size_t)16 * K;
    const __bf16* b0 = B + (size_t)(col0 + lo) * K + 8 * hi;

    bhalf16 afA[2], bfA[4], afB[2], bfB[4];
    // preload chunk 0 into set A
    afA[0] = ldfrag(a0);
    afA[1] = ldfrag(a1);
#pragma unroll
    for (int c = 0; c < 4; ++c) bfA[c] = ldfrag(b0 + (size_t)c * 16 * K);

    int kk = 0;
    for (; kk < K - 64; kk += 64) {
        // load chunk kk+32 into set B (issues as a clause; WMMAs below overlap)
        afB[0] = ldfrag(a0 + kk + 32);
        afB[1] = ldfrag(a1 + kk + 32);
#pragma unroll
        for (int c = 0; c < 4; ++c)
            bfB[c] = ldfrag(b0 + (size_t)c * 16 * K + kk + 32);
        __builtin_prefetch(a0 + kk + 96, 0, 3);
        // compute chunk kk (set A)
#pragma unroll
        for (int f = 0; f < 2; ++f)
#pragma unroll
            for (int c = 0; c < 4; ++c)
                acc[f][c] = wmma_bf16(afA[f], bfA[c], acc[f][c]);
        // load chunk kk+64 into set A
        afA[0] = ldfrag(a0 + kk + 64);
        afA[1] = ldfrag(a1 + kk + 64);
#pragma unroll
        for (int c = 0; c < 4; ++c)
            bfA[c] = ldfrag(b0 + (size_t)c * 16 * K + kk + 64);
        __builtin_prefetch(a1 + kk + 96, 0, 3);
        // compute chunk kk+32 (set B)
#pragma unroll
        for (int f = 0; f < 2; ++f)
#pragma unroll
            for (int c = 0; c < 4; ++c)
                acc[f][c] = wmma_bf16(afB[f], bfB[c], acc[f][c]);
    }
    // tail: kk == K-64; chunk kk is in set A, load final chunk K-32 into set B
    afB[0] = ldfrag(a0 + K - 32);
    afB[1] = ldfrag(a1 + K - 32);
#pragma unroll
    for (int c = 0; c < 4; ++c)
        bfB[c] = ldfrag(b0 + (size_t)c * 16 * K + K - 32);
#pragma unroll
    for (int f = 0; f < 2; ++f)
#pragma unroll
        for (int c = 0; c < 4; ++c)
            acc[f][c] = wmma_bf16(afA[f], bfA[c], acc[f][c]);
#pragma unroll
    for (int f = 0; f < 2; ++f)
#pragma unroll
        for (int c = 0; c < 4; ++c)
            acc[f][c] = wmma_bf16(afB[f], bfB[c], acc[f][c]);

    // C/D layout: element r -> row = row0 + f*16 + r + 8*hi, col = col0 + c*16 + lo
#pragma unroll
    for (int f = 0; f < 2; ++f)
#pragma unroll
        for (int c = 0; c < 4; ++c)
#pragma unroll
            for (int r = 0; r < 8; ++r)
                C[(size_t)(row0 + f * 16 + r + 8 * hi) * N + col0 + c * 16 + lo] =
                    acc[f][c][r];
}

// ---------------- gate = log_sigmoid(hs @ g_w^T) ----------------
__global__ void gate_logsig(const float* __restrict__ hs,
                            const float* __restrict__ gw,
                            float* __restrict__ gate) {
    int i = blockIdx.x * blockDim.x + threadIdx.x;
    if (i >= TSEQ * NKV) return;
    int t = i >> 3, kvh = i & 7;
    const float* hr = hs + (size_t)t * HIDDIM;
    const float* gr = gw + (size_t)kvh * HIDDIM;
    float s = 0.f;
    for (int c = 0; c < HIDDIM; ++c) s = fmaf(hr[c], gr[c], s);
    gate[i] = fminf(s, 0.f) - log1pf(__expf(-fabsf(s)));
}

// ---------------- cumulative sum over time (8 lanes, sequential) ----------------
__global__ void gate_cumsum(const float* __restrict__ g, float* __restrict__ cg) {
    int kvh = threadIdx.x;
    if (kvh >= NKV) return;
    float run = 0.f;
    for (int t = 0; t < TSEQ; ++t) {
        run += g[t * NKV + kvh];
        cg[t * NKV + kvh] = run;
    }
}

// ---------------- fused RMSNorm + RoPE (+SCALE for q) -> bf16 ----------------
// One wave per (t, head); lane handles dims {d0, d0+1, d0+64, d0+65}, d0 = 2*lane.
__global__ void __launch_bounds__(128)
normrope(const float* __restrict__ qkv, const float* __restrict__ qnw,
         const float* __restrict__ knw, const int* __restrict__ pos,
         __bf16* __restrict__ qo, __bf16* __restrict__ ko) {
    const int lane = threadIdx.x & 31;
    const int wid  = blockIdx.x * 4 + (threadIdx.x >> 5);
    const int t    = wid / (NH + NKV);
    const int hidx = wid % (NH + NKV);
    const bool isq = hidx < NH;

    const float* src = qkv + (size_t)t * QKV_N +
                       (isq ? hidx * HD : NH * HD + (hidx - NH) * HD);
    const int d0 = lane * 2;
    float x0 = src[d0], x1 = src[d0 + 1];
    float y0 = src[d0 + 64], y1 = src[d0 + 65];

    float ss = x0 * x0 + x1 * x1 + y0 * y0 + y1 * y1;
    ss += __shfl_xor(ss, 16, 32);
    ss += __shfl_xor(ss, 8, 32);
    ss += __shfl_xor(ss, 4, 32);
    ss += __shfl_xor(ss, 2, 32);
    ss += __shfl_xor(ss, 1, 32);
    const float rms = rsqrtf(ss * (1.f / HD) + RMS_EPS);

    const float* nw = isq ? qnw : knw;
    x0 *= rms * nw[d0];      x1 *= rms * nw[d0 + 1];
    y0 *= rms * nw[d0 + 64]; y1 *= rms * nw[d0 + 65];

    const float p  = (float)pos[t];
    const float sc = isq ? SCALE_Q : 1.f;
    float o0, o1, o64, o65;
    {
        float ang = p * __expf(-(float)d0 * LN_THETA_OVER_HALF);
        float cs = __cosf(ang), sn = __sinf(ang);
        o0  = (x0 * cs - y0 * sn) * sc;
        o64 = (y0 * cs + x0 * sn) * sc;
    }
    {
        float ang = p * __expf(-(float)(d0 + 1) * LN_THETA_OVER_HALF);
        float cs = __cosf(ang), sn = __sinf(ang);
        o1  = (x1 * cs - y1 * sn) * sc;
        o65 = (y1 * cs + x1 * sn) * sc;
    }
    __bf16* dst = isq ? (qo + (size_t)t * (NH * HD) + hidx * HD)
                      : (ko + (size_t)t * (NKV * HD) + (hidx - NH) * HD);
    dst[d0] = (__bf16)o0;       dst[d0 + 1] = (__bf16)o1;
    dst[d0 + 64] = (__bf16)o64; dst[d0 + 65] = (__bf16)o65;
}

// ---------------- V transpose: vT[kv*HD+d][t] = v[t][kv][d] (bf16) ----------------
__global__ void v_transpose(const float* __restrict__ qkv, __bf16* __restrict__ vT) {
    int i = blockIdx.x * 256 + threadIdx.x;
    if (i >= TSEQ * NKV * HD) return;
    int t = i / (NKV * HD);
    int rem = i - t * (NKV * HD);
    vT[(size_t)rem * TSEQ + t] =
        (__bf16)qkv[(size_t)t * QKV_N + (NH + NKV) * HD + rem];
}

// ---------------- fused causal power-retention core ----------------
// Grid (T/64, NH); 4 waves/block; wave owns a 16-row Q tile, 128 output cols.
// Per 32-key step: batched K-fragment loads (8 distinct regs -> one clause),
// 8 score WMMAs, fp32 gating/mask/square, bf16 P staged via per-wave LDS tile
// (C-layout -> A-fragment layout transform), V loads issued before the DS read,
// 8 PV WMMAs.
__global__ void __launch_bounds__(128)
retention_attn(const __bf16* __restrict__ q, const __bf16* __restrict__ k,
               const __bf16* __restrict__ vT, const float* __restrict__ cg,
               __bf16* __restrict__ attn) {
    __shared__ __align__(16) __bf16 wt[4][16][32];  // per-wave 16x32 P tile

    const int lane = threadIdx.x & 31;
    const int warp = threadIdx.x >> 5;
    const int lo = lane & 15, hi = lane >> 4;
    const int h  = blockIdx.y;
    const int kv = h >> 1;                     // rep = NH/NKV = 2
    const int qb = blockIdx.x * 64 + warp * 16;

    // Q fragments for 16 rows x 128 dims (4 K-chunks of 32)
    bhalf16 qf[4];
#pragma unroll
    for (int c = 0; c < 4; ++c) {
        const __bf16* p = q + (size_t)(qb + lo) * (NH * HD) + h * HD + c * 32 + 8 * hi;
        qf[c] = ldfrag(p);
    }
    float cgq[8];
#pragma unroll
    for (int r = 0; r < 8; ++r) cgq[r] = cg[(qb + r + 8 * hi) * NKV + kv];

    float8 oacc[8];
#pragma unroll
    for (int c = 0; c < 8; ++c) oacc[c] = f8zero();
    float wsum[8] = {0.f, 0.f, 0.f, 0.f, 0.f, 0.f, 0.f, 0.f};

    const int jend = qb + 16;                  // keys 0..qb+15 (causal)
    for (int j = 0; j < jend; j += 32) {
        // ---- batched K-fragment loads (distinct registers, single clause) ----
        bhalf16 kf0[4], kf1[4];
#pragma unroll
        for (int c = 0; c < 4; ++c) {
            const __bf16* kp0 =
                k + (size_t)(j + lo) * (NKV * HD) + kv * HD + c * 32 + 8 * hi;
            kf0[c] = ldfrag(kp0);
            kf1[c] = ldfrag(kp0 + (size_t)16 * (NKV * HD));
        }
        if (j + 32 < jend) {  // warm L2/L0 for next key block
            __builtin_prefetch(k + (size_t)(j + 32 + lo) * (NKV * HD) + kv * HD, 0, 3);
            __builtin_prefetch(k + (size_t)(j + 48 + lo) * (NKV * HD) + kv * HD, 0, 3);
        }
        const float cgk0 = cg[(j + lo) * NKV + kv];
        const float cgk1 = cg[(j + 16 + lo) * NKV + kv];

        float8 s0 = f8zero(), s1 = f8zero();
#pragma unroll
        for (int c = 0; c < 4; ++c) s0 = wmma_bf16(qf[c], kf0[c], s0);
#pragma unroll
        for (int c = 0; c < 4; ++c) s1 = wmma_bf16(qf[c], kf1[c], s1);

#pragma unroll
        for (int r = 0; r < 8; ++r) {
            const int row = qb + r + 8 * hi;
            float w0 = 0.f, w1 = 0.f;
            if (j + lo <= row) {
                float sv = s0[r];
                w0 = __expf(cgq[r] - cgk0) * sv * sv;   // P = 2
            }
            if (j + 16 + lo <= row) {
                float sv = s1[r];
                w1 = __expf(cgq[r] - cgk1) * sv * sv;
            }
            wsum[r] += w0 + w1;                          // deferred denom
            wt[warp][r + 8 * hi][lo]      = (__bf16)w0;  // C-layout -> LDS
            wt[warp][r + 8 * hi][16 + lo] = (__bf16)w1;
        }

        // ---- V fragments issued first so they overlap the DS-count wait ----
        bhalf16 vf[8];
#pragma unroll
        for (int c = 0; c < 8; ++c) {
            const __bf16* vp =
                vT + (size_t)(kv * HD + c * 16 + lo) * TSEQ + j + 8 * hi;
            vf[c] = ldfrag(vp);
        }
        // Re-read P as A-fragment (wave-coherent LDS; DS kept in order per wave)
        bhalf16 wf = ldfrag(&wt[warp][lo][8 * hi]);
#pragma unroll
        for (int c = 0; c < 8; ++c) oacc[c] = wmma_bf16(wf, vf[c], oacc[c]);
    }

    // row-sum across the 16 columns held per half-wave (xor<16 stays in half)
#pragma unroll
    for (int r = 0; r < 8; ++r) {
        float s = wsum[r];
        s += __shfl_xor(s, 1, 32);
        s += __shfl_xor(s, 2, 32);
        s += __shfl_xor(s, 4, 32);
        s += __shfl_xor(s, 8, 32);
        wsum[r] = 1.f / (s + NORM_EPS);
    }
#pragma unroll
    for (int c = 0; c < 8; ++c)
#pragma unroll
        for (int r = 0; r < 8; ++r)
            attn[(size_t)(qb + r + 8 * hi) * (NH * HD) + h * HD + c * 16 + lo] =
                (__bf16)(oacc[c][r] * wsum[r]);
}

// ---------------- host orchestration ----------------
extern "C" void kernel_launch(void* const* d_in, const int* in_sizes, int n_in,
                              void* d_out, int out_size, void* d_ws, size_t ws_size,
                              hipStream_t stream) {
    (void)in_sizes; (void)n_in; (void)out_size; (void)ws_size;
    const float* hs   = (const float*)d_in[0];
    const float* qkvw = (const float*)d_in[1];
    const float* gw   = (const float*)d_in[2];
    const float* ow   = (const float*)d_in[3];
    const float* qnw  = (const float*)d_in[4];
    const float* knw  = (const float*)d_in[5];
    const int*   pos  = (const int*)d_in[6];
    float* out = (float*)d_out;

    char* ws = (char*)d_ws;
    size_t off = 0;
    auto carve = [&](size_t bytes) -> void* {
        void* p = ws + off;
        off += (bytes + 255) & ~(size_t)255;
        return p;
    };
    __bf16* hs_b   = (__bf16*)carve((size_t)TSEQ * HIDDIM * 2);
    __bf16* qkvw_b = (__bf16*)carve((size_t)QKV_N * HIDDIM * 2);
    __bf16* ow_b   = (__bf16*)carve((size_t)HIDDIM * NH * HD * 2);
    float*  qkv_f  = (float*)carve((size_t)TSEQ * QKV_N * 4);
    __bf16* q_b    = (__bf16*)carve((size_t)TSEQ * NH * HD * 2);
    __bf16* k_b    = (__bf16*)carve((size_t)TSEQ * NKV * HD * 2);
    __bf16* vT_b   = (__bf16*)carve((size_t)NKV * HD * TSEQ * 2);
    float*  gate_f = (float*)carve((size_t)TSEQ * NKV * 4);
    float*  cg_f   = (float*)carve((size_t)TSEQ * NKV * 4);
    __bf16* attn_b = (__bf16*)carve((size_t)TSEQ * NH * HD * 2);

    // 1) pack fp32 operands to bf16 (tensor-core path)
    cvt_f32_bf16<<<(TSEQ * HIDDIM + 255) / 256, 256, 0, stream>>>(hs, hs_b, TSEQ * HIDDIM);
    cvt_f32_bf16<<<(QKV_N * HIDDIM + 255) / 256, 256, 0, stream>>>(qkvw, qkvw_b, QKV_N * HIDDIM);
    cvt_f32_bf16<<<(HIDDIM * NH * HD + 255) / 256, 256, 0, stream>>>(ow, ow_b, HIDDIM * NH * HD);

    // 2) QKV projection: [3072,2048] @ [4096,2048]^T (WMMA, pipelined)
    gemm_bf16_nt<<<dim3(QKV_N / 128, TSEQ / 128), 256, 0, stream>>>(
        hs_b, qkvw_b, qkv_f, TSEQ, QKV_N, HIDDIM);

    // 3) gate GEMV + log_sigmoid; 4) cumulative sum over time
    gate_logsig<<<(TSEQ * NKV + 255) / 256, 256, 0, stream>>>(hs, gw, gate_f);
    gate_cumsum<<<1, 32, 0, stream>>>(gate_f, cg_f);

    // 5) fused RMSNorm + RoPE (+SCALE on q) -> bf16 q/k
    normrope<<<(TSEQ * (NH + NKV)) / 4, 128, 0, stream>>>(
        qkv_f, qnw, knw, pos, q_b, k_b);

    // 6) V -> bf16 transposed [kv*HD+d][t] for contiguous B-fragments
    v_transpose<<<(TSEQ * NKV * HD + 255) / 256, 256, 0, stream>>>(qkv_f, vT_b);

    // 7) fused flash-style power-retention core (dominant WMMA work)
    retention_attn<<<dim3(TSEQ / 64, NH), 128, 0, stream>>>(
        q_b, k_b, vT_b, cg_f, attn_b);

    // 8) output projection: [3072,2048] @ [2048,2048]^T -> fp32 out (WMMA)
    gemm_bf16_nt<<<dim3(HIDDIM / 128, TSEQ / 128), 256, 0, stream>>>(
        attn_b, ow_b, out, TSEQ, HIDDIM, NH * HD);
}
